// GCN_49168785604992
// MI455X (gfx1250) — matrix operand order repaired and checked by reading, not compiled, not run on hardware
//
#include <hip/hip_runtime.h>
#include <hip/hip_bf16.h>

// ---------------- problem constants ----------------
#define BATCH    128
#define C_INN    32
#define NN       1516
#define NT       (BATCH * NN)      // 194048 nodes  (= 12128 * 16)
#define DEGC     16
#define NE       (NT * DEGC)       // 3104768 edges
#define HID      64
#define FIN      (HID * NN)        // 97024  (= 379 * 256)
#define NCLS     10
#define EPSBN    1e-5f

typedef __attribute__((ext_vector_type(16))) _Float16 v16h;
typedef __attribute__((ext_vector_type(8)))  _Float16 v8h;
typedef __attribute__((ext_vector_type(8)))  float    v8f;

// ---------------- WMMA fragment loaders (CDNA5 wave32 layouts, CDNA5 ISA 7.12.2) ----------
// A: 16x32 (MxK) f16, source already f16 row-major.
// lane L: M = M0 + (L&15); k-half offset 8 for lanes 16..31;
// elements 0..7 -> K = kb+off+0..7 ; elements 8..15 -> K = kb+off+16..23  (two b128 loads)
__device__ __forceinline__ v16h load_a_frag_h(const _Float16* __restrict__ X,
                                              int M0, int ldx, int kb, int lane) {
    const _Float16* r = X + (size_t)(M0 + (lane & 15)) * (size_t)ldx + kb + ((lane & 16) ? 8 : 0);
    v8h lo = *(const v8h*)r;
    v8h hi = *(const v8h*)(r + 16);
    v16h a;
#pragma unroll
    for (int i = 0; i < 8; ++i) { a[i] = lo[i]; a[8 + i] = hi[i]; }
    return a;
}

// B: 32x16 (KxN) f16, b(k,n) = W[(N0+n)*ldw + k] (B = W^T, W row-major [N,K], f32 source).
// lane L: N = N0 + (L&15); element i -> K = kb + (L<16 ? 0 : 8) + (i<8 ? i : i+8).
__device__ __forceinline__ v16h load_b_frag_f(const float* __restrict__ W,
                                              int N0, int ldw, int kb, int lane) {
    const float* cp = W + (size_t)(N0 + (lane & 15)) * (size_t)ldw + kb + ((lane & 16) ? 8 : 0);
    v16h b;
#pragma unroll
    for (int i = 0; i < 16; ++i) b[i] = (_Float16)cp[i < 8 ? i : i + 8];
    return b;
}

// ---------------- small utility kernels ----------------
__global__ void fill_kernel(float* __restrict__ p, float v, unsigned n) {
    unsigned i = blockIdx.x * blockDim.x + threadIdx.x;
    if (i < n) p[i] = v;
}

__global__ void deg_kernel(const int* __restrict__ col, const float* __restrict__ w,
                           float* __restrict__ deg) {
    unsigned e = blockIdx.x * blockDim.x + threadIdx.x;
    if (e < (unsigned)NE) atomicAdd(&deg[col[e]], w[e]);
}

__global__ void dinv_kernel(float* __restrict__ deg) {
    unsigned i = blockIdx.x * blockDim.x + threadIdx.x;
    if (i < (unsigned)NT) {
        float d = deg[i];
        deg[i] = (d > 0.f) ? rsqrtf(d) : 0.f;   // in-place deg -> deg^-1/2
    }
}

__global__ void norm_kernel(const int* __restrict__ row, const int* __restrict__ col,
                            const float* __restrict__ w, const float* __restrict__ dinv,
                            float* __restrict__ nrm) {
    unsigned e = blockIdx.x * blockDim.x + threadIdx.x;
    if (e < (unsigned)NE) nrm[e] = dinv[row[e]] * w[e] * dinv[col[e]];
}

// x [B, C_IN, NN] f32 -> h0 [B*NN, C_IN] f16 (GEMM A operand is consumed in f16 anyway)
__global__ void transpose_kernel(const float* __restrict__ x, _Float16* __restrict__ h0) {
    unsigned i = blockIdx.x * blockDim.x + threadIdx.x;
    if (i >= (unsigned)(NT * C_INN)) return;
    unsigned b   = i / (NN * C_INN);
    unsigned rem = i - b * (NN * C_INN);
    unsigned n   = rem / C_INN;
    unsigned c   = rem - n * C_INN;
    h0[i] = (_Float16)x[(size_t)b * C_INN * NN + (size_t)c * NN + n];
}

// ---------------- node-feature GEMM: T[NT,64] = X[NT,K] @ W[64,K]^T  (WMMA) ----------------
// grid.x = NT/32 blocks; block = 256 threads = 8 waves; wave -> (2 M-tiles x 4 N-tiles).
template <int K>
__global__ void gemm_node_kernel(const _Float16* __restrict__ X, const float* __restrict__ W,
                                 float* __restrict__ T) {
    int wave = threadIdx.x >> 5;
    int lane = threadIdx.x & 31;
    int M0 = (blockIdx.x * 2 + (wave >> 2)) * 16;
    int N0 = (wave & 3) * 16;
    v8f c = {};
#pragma unroll
    for (int kb = 0; kb < K; kb += 32) {
        v16h a = load_a_frag_h(X, M0, K, kb, lane);
        v16h b = load_b_frag_f(W, N0, K, kb, lane);
        c = __builtin_amdgcn_wmma_f32_16x16x32_f16(false, a, false, b, (short)0, c, false, false);
    }
    int n = N0 + (lane & 15);
    int mbase = M0 + ((lane & 16) ? 8 : 0);
#pragma unroll
    for (int r = 0; r < 8; ++r)
        T[(size_t)(mbase + r) * HID + n] = c[r];
}

// ---------------- edge scatter: G[col] += norm * T[row]; plus self loops ----------------
// thread t: idx = t/4 (edge or self-loop), 16-channel quarter = (t&3)*16
__global__ void aggregate_kernel(const int* __restrict__ row, const int* __restrict__ col,
                                 const float* __restrict__ nrm, const float* __restrict__ dinv,
                                 const float* __restrict__ T, float* __restrict__ G) {
    unsigned t = blockIdx.x * blockDim.x + threadIdx.x;
    unsigned idx = t >> 2;
    unsigned q   = (t & 3u) << 4;
    if (idx >= (unsigned)(NE + NT)) return;
    int rsrc, cdst; float w;
    if (idx < (unsigned)NE) {
        rsrc = row[idx]; cdst = col[idx]; w = nrm[idx];
    } else {
        rsrc = cdst = (int)(idx - NE);
        float d = dinv[rsrc]; w = d * d;          // self loop: weight 1 * dinv^2
    }
    const float* sp = T + (size_t)rsrc * HID + q;
    float*       dp = G + (size_t)cdst * HID + q;
    __builtin_prefetch(dp, 0, 3);                 // pull scatter target line (global_prefetch_b8)
#pragma unroll
    for (int i = 0; i < 16; i += 4) {
        float4 v = *(const float4*)(sp + i);
        atomicAdd(dp + i + 0, w * v.x);
        atomicAdd(dp + i + 1, w * v.y);
        atomicAdd(dp + i + 2, w * v.z);
        atomicAdd(dp + i + 3, w * v.w);
    }
}

// ---------------- BN stats over 194048 rows x 64 ch (sums[0:64]=sum, sums[64:128]=sumsq) ----
__global__ void bn_stats_kernel(const float* __restrict__ G, float* __restrict__ sums) {
    __shared__ float ls[128];
    if (threadIdx.x < 128) ls[threadIdx.x] = 0.f;
    __syncthreads();
    unsigned tid    = blockIdx.x * blockDim.x + threadIdx.x;
    unsigned stride = gridDim.x * blockDim.x;          // multiple of 64 -> channel fixed
    int c = tid & 63;
    float s = 0.f, qq = 0.f;
    for (unsigned i = tid; i < (unsigned)(NT * HID); i += stride) {
        float v = G[i]; s += v; qq += v * v;
    }
    atomicAdd(&ls[c], s);
    atomicAdd(&ls[64 + c], qq);
    __syncthreads();
    if (threadIdx.x < 128) atomicAdd(&sums[threadIdx.x], ls[threadIdx.x]);
}

// BN + ReLU (conv bias cancels inside BN exactly); writes next-layer GEMM input in f16
__global__ void bn_apply_kernel(const float* __restrict__ G, const float* __restrict__ sums,
                                const float* __restrict__ g, const float* __restrict__ be,
                                _Float16* __restrict__ Y) {
    unsigned i = blockIdx.x * blockDim.x + threadIdx.x;
    if (i >= (unsigned)(NT * HID)) return;
    int c = i & 63;
    const float invN = 1.f / (float)NT;
    float mu  = sums[c] * invN;
    float var = sums[64 + c] * invN - mu * mu;
    float y = g[c] * (G[i] - mu) * rsqrtf(var + EPSBN) + be[c];
    Y[i] = (_Float16)(y > 0.f ? y : 0.f);
}

// ---------------- big linear: acc[128,128] += H[128,97024] @ Wl1[128,97024]^T (split-K WMMA) --
// grid = (379 K-slices, 8 N-tile groups); block 256 = 8 waves; wave w -> M-tile w.
__global__ void gemm_lin1_kernel(const _Float16* __restrict__ H, const float* __restrict__ Wl1,
                                 float* __restrict__ acc) {
    int wave = threadIdx.x >> 5;
    int lane = threadIdx.x & 31;
    int M0 = wave * 16;
    int N0 = blockIdx.y * 16;
    int k0 = blockIdx.x * 256;
    v8f c = {};
#pragma unroll
    for (int kb = 0; kb < 256; kb += 32) {
        v16h a = load_a_frag_h(H,   M0, FIN, k0 + kb, lane);
        v16h b = load_b_frag_f(Wl1, N0, FIN, k0 + kb, lane);
        c = __builtin_amdgcn_wmma_f32_16x16x32_f16(false, a, false, b, (short)0, c, false, false);
    }
    int n = N0 + (lane & 15);
    int mbase = M0 + ((lane & 16) ? 8 : 0);
#pragma unroll
    for (int r = 0; r < 8; ++r)
        atomicAdd(&acc[(size_t)(mbase + r) * 128 + n], c[r]);
}

// ---------------- batch BN stats for [128,128] ----------------
__global__ void bn2_stats_kernel(const float* __restrict__ acc,
                                 float* __restrict__ mu2, float* __restrict__ var2) {
    int j = threadIdx.x;          // 128 threads
    float s = 0.f, qq = 0.f;
    for (int b = 0; b < BATCH; ++b) {
        float v = acc[(size_t)b * 128 + j];
        s += v; qq += v * v;
    }
    float mu = s / (float)BATCH;
    mu2[j]  = mu;
    var2[j] = qq / (float)BATCH - mu * mu;
}

// ---------------- BN + ReLU + final [128]->[10] ----------------
__global__ void final_kernel(const float* __restrict__ acc, const float* __restrict__ mu2,
                             const float* __restrict__ var2, const float* __restrict__ gl1,
                             const float* __restrict__ bel1, const float* __restrict__ Wl3,
                             const float* __restrict__ bl3, float* __restrict__ out) {
    __shared__ float ys[128];
    int b = blockIdx.x, j = threadIdx.x;   // 128 threads
    float v = acc[(size_t)b * 128 + j];
    float y = gl1[j] * (v - mu2[j]) * rsqrtf(var2[j] + EPSBN) + bel1[j];
    ys[j] = y > 0.f ? y : 0.f;
    __syncthreads();
    if (j < NCLS) {
        float s = bl3[j];
#pragma unroll 8
        for (int t = 0; t < 128; ++t) s += ys[t] * Wl3[(size_t)j * 128 + t];
        out[(size_t)b * NCLS + j] = s;
    }
}

// ---------------- launch ----------------
static inline unsigned cdiv(unsigned a, unsigned b) { return (a + b - 1) / b; }

extern "C" void kernel_launch(void* const* d_in, const int* in_sizes, int n_in,
                              void* d_out, int out_size, void* d_ws, size_t ws_size,
                              hipStream_t stream) {
    (void)in_sizes; (void)n_in; (void)out_size; (void)ws_size;
    const float* x    = (const float*)d_in[0];
    const int*   ei   = (const int*)  d_in[1];     // [2, NE]: row = ei, col = ei+NE
    const float* ew   = (const float*)d_in[2];
    const float* W1   = (const float*)d_in[3];
    const float* W2   = (const float*)d_in[5];
    const float* W3   = (const float*)d_in[7];
    const float* g1   = (const float*)d_in[9];
    const float* be1  = (const float*)d_in[10];
    const float* g2   = (const float*)d_in[11];
    const float* be2  = (const float*)d_in[12];
    const float* g3   = (const float*)d_in[13];
    const float* be3  = (const float*)d_in[14];
    const float* Wl1  = (const float*)d_in[15];
    const float* gl1  = (const float*)d_in[17];
    const float* bel1 = (const float*)d_in[18];
    const float* Wl3  = (const float*)d_in[19];
    const float* bl3  = (const float*)d_in[20];
    // b1/b2/b3/bl1 are mathematically cancelled by the following BatchNorm.

    // workspace layout
    _Float16* Xb = (_Float16*)d_ws;                       // NT*64 f16: GEMM A operand
    float* Tb   = (float*)(Xb + (size_t)NT * HID);        // NT*64 f32: GEMM output
    float* Gb   = Tb  + (size_t)NT * HID;                 // NT*64 f32: aggregation output
    float* nrm  = Gb  + (size_t)NT * HID;                 // NE        edge norms
    float* deg  = nrm + (size_t)NE;                       // NT        degree -> dinv (in place)
    float* sums = deg + (size_t)NT;                       // 128       BN sums/sumsq
    float* acc  = sums + 128;                             // 128*128   head accumulator
    float* mu2  = acc + 128 * 128;                        // 128
    float* var2 = mu2 + 128;                              // 128

    const int* row = ei;
    const int* col = ei + NE;

    // gcn_norm
    fill_kernel<<<cdiv(NT, 256), 256, 0, stream>>>(deg, 1.0f, NT);   // self-loop weight 1
    deg_kernel <<<cdiv(NE, 256), 256, 0, stream>>>(col, ew, deg);
    dinv_kernel<<<cdiv(NT, 256), 256, 0, stream>>>(deg);
    norm_kernel<<<cdiv(NE, 256), 256, 0, stream>>>(row, col, ew, deg, nrm);

    // [B,C,N] -> [B*N, C] in f16
    transpose_kernel<<<cdiv(NT * C_INN, 256), 256, 0, stream>>>(x, Xb);

    const unsigned aggGrid = cdiv((NE + NT) * 4u, 256);
    const unsigned elemGrid = cdiv(NT * HID, 256);

    // ---- layer 1 (K = 32) ----
    gemm_node_kernel<C_INN><<<NT / 32, 256, 0, stream>>>(Xb, W1, Tb);
    fill_kernel<<<elemGrid, 256, 0, stream>>>(Gb, 0.f, NT * HID);
    aggregate_kernel<<<aggGrid, 256, 0, stream>>>(row, col, nrm, deg, Tb, Gb);
    fill_kernel<<<1, 256, 0, stream>>>(sums, 0.f, 128);
    bn_stats_kernel<<<4096, 256, 0, stream>>>(Gb, sums);
    bn_apply_kernel<<<elemGrid, 256, 0, stream>>>(Gb, sums, g1, be1, Xb);

    // ---- layer 2 (K = 64) ----
    gemm_node_kernel<HID><<<NT / 32, 256, 0, stream>>>(Xb, W2, Tb);
    fill_kernel<<<elemGrid, 256, 0, stream>>>(Gb, 0.f, NT * HID);
    aggregate_kernel<<<aggGrid, 256, 0, stream>>>(row, col, nrm, deg, Tb, Gb);
    fill_kernel<<<1, 256, 0, stream>>>(sums, 0.f, 128);
    bn_stats_kernel<<<4096, 256, 0, stream>>>(Gb, sums);
    bn_apply_kernel<<<elemGrid, 256, 0, stream>>>(Gb, sums, g2, be2, Xb);

    // ---- layer 3 (K = 64) ----
    gemm_node_kernel<HID><<<NT / 32, 256, 0, stream>>>(Xb, W3, Tb);
    fill_kernel<<<elemGrid, 256, 0, stream>>>(Gb, 0.f, NT * HID);
    aggregate_kernel<<<aggGrid, 256, 0, stream>>>(row, col, nrm, deg, Tb, Gb);
    fill_kernel<<<1, 256, 0, stream>>>(sums, 0.f, 128);
    bn_stats_kernel<<<4096, 256, 0, stream>>>(Gb, sums);
    bn_apply_kernel<<<elemGrid, 256, 0, stream>>>(Gb, sums, g3, be3, Xb);

    // ---- head: split-K WMMA into f32 accumulator ----
    fill_kernel<<<cdiv(128 * 128, 256), 256, 0, stream>>>(acc, 0.f, 128 * 128);
    gemm_lin1_kernel<<<dim3(FIN / 256, 8), 256, 0, stream>>>(Xb, Wl1, acc);
    bn2_stats_kernel<<<1, 128, 0, stream>>>(acc, mu2, var2);
    final_kernel<<<BATCH, 128, 0, stream>>>(acc, mu2, var2, gl1, bel1, Wl3, bl3, (float*)d_out);
}
